// MambaBlock_49512382988927
// MI455X (gfx1250) — matrix operand order repaired and checked by reading, not compiled, not run on hardware
//
#include <hip/hip_runtime.h>
#include <hip/hip_bf16.h>

typedef __attribute__((ext_vector_type(16))) __bf16    v16bf;
typedef __attribute__((ext_vector_type(8)))  float     v8f;
typedef __attribute__((ext_vector_type(4)))  unsigned  v4u;
typedef __attribute__((ext_vector_type(8)))  unsigned  v8u;

#define THREADS 256
#define BM 128
#define BN 128
#define BK 32
#define LDSS 48   // padded LDS row stride (bf16 elems): 96B rows = 64B data + 32B pad

static __device__ __forceinline__ float siluf(float v) {
    return v * (1.0f / (1.0f + __expf(-v)));
}
static __device__ __forceinline__ float softplusf(float v) {
    return (v > 20.0f) ? v : logf(1.0f + __expf(v));
}

union FragU { uint4 u[2]; v16bf v; };

static __device__ __forceinline__ void async_b128(unsigned lds_addr,
                                                  const void* gaddr) {
    // CDNA5 async copy: 16B / lane, global -> LDS, tracked by ASYNCcnt.
    asm volatile("global_load_async_to_lds_b128 %0, %1, off"
                 :: "v"(lds_addr), "v"((unsigned long long)gaddr)
                 : "memory");
}
static __device__ __forceinline__ void wait_async0() {
    asm volatile("s_wait_asynccnt 0x0" ::: "memory");
}

// ---------------------------------------------------------------------------
// TDM: DMA one [128 rows x 32 bf16] tile (row stride = K elems) into LDS with
// hardware row padding 64B data + 32B pad  ->  LDSS=48-elem LDS rows.
// 2D tensor: descriptor groups 0 and 1 only (VADDR2/3 omitted -> NULL).
// ---------------------------------------------------------------------------
static __device__ __forceinline__ void tdm_load_tile_128x32(
    unsigned lds_byte_addr, const void* gaddr, unsigned long long k_stride_elems)
{
    unsigned long long ga = (unsigned long long)gaddr;
    v4u g0;
    g0[0] = 1u;                                   // count=1 (valid), user mode
    g0[1] = lds_byte_addr;                        // lds_addr
    g0[2] = (unsigned)ga;                         // global_addr[31:0]
    g0[3] = (unsigned)((ga >> 32) & 0x01FFFFFFu)  // global_addr[56:32]
          | (2u << 30);                           // type = 2 ("image")
    v8u g1;
    g1[0] = (1u << 16)                            // data_size = 1 -> 2 bytes
          | (1u << 20)                            // pad_enable
          | (3u << 22)                            // pad_interval: 16 DWORDs (64B)
          | (7u << 25);                           // pad_amount: 8 DWORDs (32B)
    g1[1] = (32u & 0xFFFFu) << 16;                // tensor_dim0[15:0] = 32
    g1[2] = 0u | (128u << 16);                    // dim0[31:16]=0, tensor_dim1[15:0]=128
    g1[3] = 0u | (32u << 16);                     // dim1[31:16]=0, tile_dim0=32
    g1[4] = 128u;                                 // tile_dim1=128, tile_dim2=0
    g1[5] = (unsigned)k_stride_elems;             // tensor_dim0_stride[31:0]
    g1[6] = (unsigned)((k_stride_elems >> 32) & 0xFFFFu); // stride[47:32], dim1_stride lo=0
    g1[7] = 0u;
    asm volatile("tensor_load_to_lds %0, %1"
                 :: "s"(g0), "s"(g1)
                 : "memory");
}

// ---------------------------------------------------------------------------
// Linear GEMM: C[M,N] = act( A[M,K] @ W[N,K]^T + bias[n] ) [* aux]
// Double-buffered LDS; tiles staged by the Tensor Data Mover (wave 0 issues
// 2 descriptors / K-step), TENSORcnt-synchronized; 8 WMMA / wave / K-step.
// ---------------------------------------------------------------------------
template<int ACT, bool MULAUX, bool OUTF, bool OUTB>
__global__ __launch_bounds__(THREADS)
void gemm_lin_wmma(const __bf16* __restrict__ A,
                   const __bf16* __restrict__ Bw,
                   const float*  __restrict__ bias,
                   const float*  __restrict__ aux,
                   float*        __restrict__ outf,
                   __bf16*       __restrict__ outb,
                   int M, int N, int K)
{
    __shared__ __bf16 As[2][BM][LDSS];   // 2 * 12288 B
    __shared__ __bf16 Bs[2][BN][LDSS];   // 2 * 12288 B

    const int tid  = threadIdx.x;
    const int lane = tid & 31;
    const int wv   = tid >> 5;
    const int wm   = wv >> 2;           // 0..1
    const int wn   = wv & 3;            // 0..3
    const int lm   = lane & 15;
    const int lh   = lane >> 4;

    const int m0 = blockIdx.y * BM;
    const int n0 = blockIdx.x * BN;

    const unsigned ldsA = (unsigned)(unsigned long long)(void*)&As[0][0][0];
    const unsigned ldsB = (unsigned)(unsigned long long)(void*)&Bs[0][0][0];

    // Uniform scalar guard: TDM ignores EXEC, so gate with a wave-uniform
    // SGPR condition (only wave 0 issues descriptors).
    const bool tdm_wave = (__builtin_amdgcn_readfirstlane(tid) >> 5) == 0;

    v8f acc[4][2];
    #pragma unroll
    for (int i = 0; i < 4; ++i)
        #pragma unroll
        for (int j = 0; j < 2; ++j)
            #pragma unroll
            for (int e = 0; e < 8; ++e) acc[i][j][e] = 0.0f;

    auto stage = [&](int kb, int buf) {
        if (tdm_wave) {
            tdm_load_tile_128x32(ldsA + (unsigned)(buf * BM * LDSS * 2),
                                 &A[(size_t)m0 * K + kb], (unsigned long long)K);
            tdm_load_tile_128x32(ldsB + (unsigned)(buf * BN * LDSS * 2),
                                 &Bw[(size_t)n0 * K + kb], (unsigned long long)K);
        }
    };

    stage(0, 0);
    __builtin_amdgcn_s_wait_tensorcnt(0);
    __syncthreads();

    int buf = 0;
    for (int kb = 0; kb < K; kb += BK, buf ^= 1) {
        if (kb + BK < K) stage(kb + BK, buf ^ 1);   // DMA next tile into other buffer

        FragU fa[4], fb[2];
        #pragma unroll
        for (int im = 0; im < 4; ++im) {
            int m = wm * 64 + im * 16 + lm;
            fa[im].u[0] = *(const uint4*)&As[buf][m][lh * 8];
            fa[im].u[1] = *(const uint4*)&As[buf][m][lh * 8 + 16];
        }
        #pragma unroll
        for (int in = 0; in < 2; ++in) {
            int n = wn * 32 + in * 16 + lm;
            fb[in].u[0] = *(const uint4*)&Bs[buf][n][lh * 8];
            fb[in].u[1] = *(const uint4*)&Bs[buf][n][lh * 8 + 16];
        }
        #pragma unroll
        for (int im = 0; im < 4; ++im)
            #pragma unroll
            for (int in = 0; in < 2; ++in)
                acc[im][in] = __builtin_amdgcn_wmma_f32_16x16x32_bf16(
                    false, fa[im].v, false, fb[in].v,
                    (short)0, acc[im][in], false, false);

        __builtin_amdgcn_s_wait_tensorcnt(0);  // next tile fully in LDS (wave 0)
        __syncthreads();                       // publish; buf readers all done
    }

    #pragma unroll
    for (int im = 0; im < 4; ++im) {
        #pragma unroll
        for (int in = 0; in < 2; ++in) {
            int ng = n0 + wn * 32 + in * 16 + lm;
            #pragma unroll
            for (int r = 0; r < 8; ++r) {
                int m = m0 + wm * 64 + im * 16 + r + 8 * lh;
                float v = acc[im][in][r] + bias[ng];
                if (ACT == 1) v = siluf(v);
                else if (ACT == 2) v = softplusf(v);
                size_t o = (size_t)m * N + ng;
                if (MULAUX) v *= aux[o];
                if (OUTF) outf[o] = v;
                if (OUTB) outb[o] = (__bf16)v;
            }
        }
    }
}

// ---------------------------------------------------------------------------
// Conv-as-GEMM (implicit im2col): out[o,w] = silu( sum_{i,kk} Wc[o,i*3+kk] *
// xproj[i, w+kk-1] + bconv[o] ), M=L(o), N=2D(w), K=3L. A staged via async
// copy, B gathered (two-phase: registers then LDS).
// ---------------------------------------------------------------------------
__global__ __launch_bounds__(THREADS)
void gemm_conv_wmma(const __bf16* __restrict__ A,      // Wconv bf16 [L, 3L]
                    const __bf16* __restrict__ Xp,     // x_proj[batch] [L, ldb]
                    const float*  __restrict__ bias,   // bconv (per row)
                    __bf16*       __restrict__ outb,
                    int M, int N, int K, int ldb)
{
    __shared__ __bf16 As[BM][LDSS];
    __shared__ __bf16 Bs[BN][LDSS];

    const int tid  = threadIdx.x;
    const int wv   = tid >> 5;
    const int lane = tid & 31;
    const int wm   = wv >> 2;
    const int wn   = wv & 3;
    const int lm   = lane & 15;
    const int lh   = lane >> 4;

    const int m0 = blockIdx.y * BM;
    const int n0 = blockIdx.x * BN;

    const unsigned ldsA = (unsigned)(unsigned long long)(void*)&As[0][0];

    v8f acc[4][2];
    #pragma unroll
    for (int i = 0; i < 4; ++i)
        #pragma unroll
        for (int j = 0; j < 2; ++j)
            #pragma unroll
            for (int e = 0; e < 8; ++e) acc[i][j][e] = 0.0f;

    for (int kb = 0; kb < K; kb += BK) {
        __syncthreads();   // LDS free (prev iter readers done)

        #pragma unroll
        for (int it = 0; it < 2; ++it) {
            int idx = tid + it * THREADS;
            int r = idx >> 2;
            int c = (idx & 3) * 8;
            async_b128(ldsA + (unsigned)((r * LDSS + c) * 2),
                       &A[(size_t)(m0 + r) * K + kb + c]);
        }
        // phase 1: gather B values into registers (loads batch under one wait)
        __bf16 tmp[16];
        #pragma unroll
        for (int it = 0; it < 16; ++it) {
            int idx = tid + it * THREADS; // 0..4095
            int n  = idx & 127;
            int kg = kb + (idx >> 7);
            int i  = kg / 3;
            int kk = kg - i * 3;
            int wcol = n0 + n + kk - 1;
            __bf16 v = (__bf16)0.0f;
            if ((unsigned)wcol < (unsigned)ldb)
                v = Xp[(size_t)i * ldb + wcol];
            tmp[it] = v;
        }
        // phase 2: scatter to LDS
        #pragma unroll
        for (int it = 0; it < 16; ++it) {
            int idx = tid + it * THREADS;
            Bs[idx & 127][idx >> 7] = tmp[it];
        }
        wait_async0();
        __syncthreads();

        FragU fa[4], fb[2];
        #pragma unroll
        for (int im = 0; im < 4; ++im) {
            int m = wm * 64 + im * 16 + lm;
            fa[im].u[0] = *(const uint4*)&As[m][lh * 8];
            fa[im].u[1] = *(const uint4*)&As[m][lh * 8 + 16];
        }
        #pragma unroll
        for (int in = 0; in < 2; ++in) {
            int n = wn * 32 + in * 16 + lm;
            fb[in].u[0] = *(const uint4*)&Bs[n][lh * 8];
            fb[in].u[1] = *(const uint4*)&Bs[n][lh * 8 + 16];
        }
        #pragma unroll
        for (int im = 0; im < 4; ++im)
            #pragma unroll
            for (int in = 0; in < 2; ++in)
                acc[im][in] = __builtin_amdgcn_wmma_f32_16x16x32_bf16(
                    false, fa[im].v, false, fb[in].v,
                    (short)0, acc[im][in], false, false);
    }

    #pragma unroll
    for (int im = 0; im < 4; ++im) {
        #pragma unroll
        for (int in = 0; in < 2; ++in) {
            int ng = n0 + wn * 32 + in * 16 + lm;
            #pragma unroll
            for (int r = 0; r < 8; ++r) {
                int m = m0 + wm * 64 + im * 16 + r + 8 * lh;
                float v = siluf(acc[im][in][r] + bias[m]);
                outb[(size_t)m * N + ng] = (__bf16)v;
            }
        }
    }
}

__global__ void f32_to_bf16_kernel(const float* __restrict__ in,
                                   __bf16* __restrict__ out, size_t n) {
    size_t i = (size_t)blockIdx.x * blockDim.x + threadIdx.x;
    size_t stride = (size_t)gridDim.x * blockDim.x;
    for (; i < n; i += stride) out[i] = (__bf16)in[i];
}

__global__ __launch_bounds__(256)
void rmsnorm_kernel(const float* __restrict__ x, const float* __restrict__ nw,
                    __bf16* __restrict__ xn, int D) {
    int row = blockIdx.x, t = threadIdx.x;
    const float* xr = x + (size_t)row * D;
    float ss = 0.0f;
    for (int d = t; d < D; d += 256) { float v = xr[d]; ss += v * v; }
    __shared__ float red[256];
    red[t] = ss; __syncthreads();
    for (int off = 128; off > 0; off >>= 1) {
        if (t < off) red[t] += red[t + off];
        __syncthreads();
    }
    float inv = rsqrtf(red[0] / (float)D + 1e-5f);
    for (int d = t; d < D; d += 256)
        xn[(size_t)row * D + d] = (__bf16)(xr[d] * inv * nw[d]);
}

// s[m] = sum_n (xco[m,:]·W2[n,:] + b2[n]) * (xco[m,:]·W3[n,:] + b3[n])
__global__ __launch_bounds__(256)
void s6_scalar_kernel(const float* __restrict__ xco,
                      const float* __restrict__ W2, const float* __restrict__ b2,
                      const float* __restrict__ W3, const float* __restrict__ b3,
                      float* __restrict__ s, int K) {
    int m = blockIdx.x, t = threadIdx.x;
    int n = t >> 4, l = t & 15;
    const float* row = xco + (size_t)m * K;
    float pb = 0.0f, pc = 0.0f;
    for (int k = l; k < K; k += 16) {
        float xv = row[k];
        pb += xv * W2[(size_t)n * K + k];
        pc += xv * W3[(size_t)n * K + k];
    }
    __shared__ float sb[16][16], sc[16][16], sn[16];
    sb[n][l] = pb; sc[n][l] = pc;
    __syncthreads();
    if (l == 0) {
        float bb = 0.0f, cc = 0.0f;
        for (int j = 0; j < 16; ++j) { bb += sb[n][j]; cc += sc[n][j]; }
        sn[n] = (bb + b2[n]) * (cc + b3[n]);
    }
    __syncthreads();
    if (t == 0) {
        float a = 0.0f;
        for (int j = 0; j < 16; ++j) a += sn[j];
        s[m] = a;
    }
}

__global__ void ssm_act_kernel(const float* __restrict__ xco,
                               float* __restrict__ dio,
                               const float* __restrict__ s, size_t total) {
    size_t i = (size_t)blockIdx.x * blockDim.x + threadIdx.x;
    if (i >= total) return;
    float v = xco[i] * dio[i] * s[i >> 11];   // row = i / 2048
    dio[i] = siluf(v);
}

extern "C" void kernel_launch(void* const* d_in, const int* in_sizes, int n_in,
                              void* d_out, int out_size, void* d_ws, size_t ws_size,
                              hipStream_t stream) {
    const int Bb = 2, L = 1024, Dm = 1024;
    const int D2 = 2 * Dm;                 // 2048
    const int Mrows = Bb * L;              // 2048

    const float* x     = (const float*)d_in[0];
    const float* nw    = (const float*)d_in[1];
    const float* Wi    = (const float*)d_in[2];
    const float* bi    = (const float*)d_in[3];
    const float* Wconv = (const float*)d_in[4];
    const float* bconv = (const float*)d_in[5];
    const float* Wcl   = (const float*)d_in[6];
    const float* bcl   = (const float*)d_in[7];
    const float* W1    = (const float*)d_in[8];
    const float* b1    = (const float*)d_in[9];
    const float* W2    = (const float*)d_in[10];
    const float* b2    = (const float*)d_in[11];
    const float* W3    = (const float*)d_in[12];
    const float* b3    = (const float*)d_in[13];
    /* d_in[14] = A : provably unused (h0 == 0) */
    const float* WD    = (const float*)d_in[15];
    const float* bD    = (const float*)d_in[16];
    const float* Wo    = (const float*)d_in[17];
    const float* bo    = (const float*)d_in[18];
    float* out = (float*)d_out;

    char* ws = (char*)d_ws;
    size_t off = 0;
    auto alloc = [&](size_t bytes) -> void* {
        void* p = ws + off;
        off = (off + bytes + 255) & ~(size_t)255;
        return p;
    };
    __bf16* Wi_bf     = (__bf16*)alloc((size_t)D2 * Dm * 2);
    __bf16* Wconv_bf  = (__bf16*)alloc((size_t)L * L * 3 * 2);
    __bf16* Wcl_bf    = (__bf16*)alloc((size_t)D2 * D2 * 2);
    __bf16* W1_bf     = (__bf16*)alloc((size_t)D2 * D2 * 2);
    __bf16* WD_bf     = (__bf16*)alloc((size_t)D2 * Dm * 2);
    __bf16* Wo_bf     = (__bf16*)alloc((size_t)Dm * D2 * 2);
    __bf16* xn_bf     = (__bf16*)alloc((size_t)Mrows * Dm * 2);
    __bf16* xproj_bf  = (__bf16*)alloc((size_t)Mrows * D2 * 2);
    __bf16* xconvs_bf = (__bf16*)alloc((size_t)Mrows * D2 * 2);
    float*  xco_f     = (float*) alloc((size_t)Mrows * D2 * 4);
    __bf16* xco_bf    = (__bf16*)alloc((size_t)Mrows * D2 * 2);
    float*  delta_f   = (float*) alloc((size_t)Mrows * D2 * 4);   // reused as x_act
    float*  s_f       = (float*) alloc((size_t)Mrows * 4);
    __bf16* y_bf      = xproj_bf;  // reuse: x_proj dead after conv GEMM

    auto cvt = [&](const float* src, __bf16* dst, size_t n) {
        f32_to_bf16_kernel<<<2048, 256, 0, stream>>>(src, dst, n);
    };
    cvt(Wi,    Wi_bf,    (size_t)D2 * Dm);
    cvt(Wconv, Wconv_bf, (size_t)L * L * 3);
    cvt(Wcl,   Wcl_bf,   (size_t)D2 * D2);
    cvt(W1,    W1_bf,    (size_t)D2 * D2);
    cvt(WD,    WD_bf,    (size_t)D2 * Dm);
    cvt(Wo,    Wo_bf,    (size_t)Dm * D2);

    // 1) RMSNorm -> xn (bf16)
    rmsnorm_kernel<<<Mrows, 256, 0, stream>>>(x, nw, xn_bf, Dm);

    // 2) x_proj = xn @ Wi^T + bi
    gemm_lin_wmma<0, false, false, true>
        <<<dim3(D2 / BN, Mrows / BM), THREADS, 0, stream>>>(
            xn_bf, Wi_bf, bi, nullptr, nullptr, xproj_bf, Mrows, D2, Dm);

    // 3) conv as implicit-im2col GEMM per batch: silu(conv + bconv[o]) -> bf16
    for (int b = 0; b < Bb; ++b)
        gemm_conv_wmma<<<dim3(D2 / BN, L / BM), THREADS, 0, stream>>>(
            Wconv_bf, xproj_bf + (size_t)b * L * D2, bconv,
            xconvs_bf + (size_t)b * L * D2, L, D2, L * 3, D2);

    // 4) x_conv_out = silu(x_conv) @ Wcl^T + bcl  (f32 + bf16 copies)
    gemm_lin_wmma<0, false, true, true>
        <<<dim3(D2 / BN, Mrows / BM), THREADS, 0, stream>>>(
            xconvs_bf, Wcl_bf, bcl, nullptr, xco_f, xco_bf, Mrows, D2, D2);

    // 5) delta = softplus(x_conv_out @ W1^T + b1)
    gemm_lin_wmma<2, false, true, false>
        <<<dim3(D2 / BN, Mrows / BM), THREADS, 0, stream>>>(
            xco_bf, W1_bf, b1, nullptr, delta_f, nullptr, Mrows, D2, D2);

    // 6) s[m] = (xco·W2^T + b2)·(xco·W3^T + b3)  (einsum collapses over n)
    s6_scalar_kernel<<<Mrows, 256, 0, stream>>>(xco_f, W2, b2, W3, b3, s_f, D2);

    // 7) x_act = silu(xco * delta * s)  (in-place over delta buffer)
    {
        size_t total = (size_t)Mrows * D2;
        ssm_act_kernel<<<(unsigned)((total + 255) / 256), 256, 0, stream>>>(
            xco_f, delta_f, s_f, total);
    }

    // 8) y = x_act * silu(xn @ WD^T + bD)  -> bf16
    gemm_lin_wmma<1, true, false, true>
        <<<dim3(D2 / BN, Mrows / BM), THREADS, 0, stream>>>(
            xn_bf, WD_bf, bD, delta_f, nullptr, y_bf, Mrows, D2, Dm);

    // 9) out = y @ Wo^T + bo  (f32, final)
    gemm_lin_wmma<0, false, true, false>
        <<<dim3(Dm / BN, Mrows / BM), THREADS, 0, stream>>>(
            y_bf, Wo_bf, bo, nullptr, out, nullptr, Mrows, Dm, D2);

    (void)in_sizes; (void)n_in; (void)out_size; (void)ws_size;
}
